// RegionProposalNetwork_33629593928101
// MI455X (gfx1250) — matrix operand order repaired and checked by reading, not compiled, not run on hardware
//
#include <hip/hip_runtime.h>

typedef __attribute__((ext_vector_type(16))) _Float16 v16h;
typedef __attribute__((ext_vector_type(8)))  _Float16 v8h;
typedef __attribute__((ext_vector_type(8)))  float    v8f;

#define N_ANCH 258048

__constant__ int   c_lvlOff[3]    = {0, 196608, 245760};
__constant__ int   c_lvlCnt[3]    = {196608, 49152, 12288};
__constant__ int   c_lvlW[3]      = {256, 128, 64};
__constant__ int   c_lvlLog2W[3]  = {8, 7, 6};
__constant__ float c_lvlStride[3] = {8.f, 16.f, 32.f};
// round(base/2) with banker's rounding, per reference _cell_anchors
__constant__ float c_wa[3][3] = {{22.f, 32.f, 46.f}, {46.f, 64.f, 90.f}, {90.f, 128.f, 182.f}};
__constant__ float c_ha[3][3] = {{46.f, 32.f, 22.f}, {90.f, 64.f, 46.f}, {182.f, 128.f, 90.f}};

__device__ __forceinline__ unsigned sortKey(float f) {
  unsigned u = __float_as_uint(f);
  return (u & 0x80000000u) ? ~u : (u | 0x80000000u);
}

// ---------------- prep: weights -> f16 WMMA-friendly layouts ----------------
__global__ void k_prep(const float* __restrict__ cw, const float* __restrict__ clsw,
                       const float* __restrict__ bbxw, _Float16* __restrict__ Bprep,
                       _Float16* __restrict__ Bh) {
  const int tid = blockIdx.x * blockDim.x + threadIdx.x;
  const int stride = gridDim.x * blockDim.x;
  // Bprep[tap][n][k] = conv_w[n][k][dh][dw]
  for (int i = tid; i < 9 * 4096; i += stride) {
    const int tap = i >> 12, rem = i & 4095;
    const int n = rem >> 6, k = rem & 63;
    const int dh = tap / 3, dw = tap % 3;
    Bprep[i] = (_Float16)cw[((n * 64 + k) * 3 + dh) * 3 + dw];
  }
  // Bh[n][k]: cols 0..2 = cls, 3..14 = bbx, 15 = 0
  for (int i = tid; i < 1024; i += stride) {
    const int n = i >> 6, k = i & 63;
    float v = 0.f;
    if (n < 3) v = clsw[n * 64 + k];
    else if (n < 15) v = bbxw[(n - 3) * 64 + k];
    Bh[i] = (_Float16)v;
  }
}

// ---------------- NCHW f32 -> NHWC f16 ----------------
__global__ void k_nhwc(const float* __restrict__ f, _Float16* __restrict__ xo,
                       int HW, int log2HW, int total) {
  for (int o = blockIdx.x * blockDim.x + threadIdx.x; o < total; o += gridDim.x * blockDim.x) {
    const int c = o & 63;
    const int rem = o >> 6;            // img*HW + pix
    const int pix = rem & (HW - 1);
    const int img = rem >> log2HW;
    xo[o] = (_Float16)f[(size_t)(img * 64 + c) * HW + pix];
  }
}

// ---------------- fused 3x3 conv + 1x1 heads, all WMMA ----------------
__global__ __launch_bounds__(256) void k_conv_head(
    const _Float16* __restrict__ xLvl, const _Float16* __restrict__ Bprep,
    const _Float16* __restrict__ Bh, const float* __restrict__ conv_b,
    const float* __restrict__ cls_b, const float* __restrict__ bbx_b,
    float* __restrict__ scores, float* __restrict__ deltas,
    int W, int log2W, int HW, int lvlOff) {
  __shared__ alignas(32) _Float16 yTile[8][1024];   // 8 waves x (16 px * 64 ch)
  const int lane = threadIdx.x & 31;
  const int wave = threadIdx.x >> 5;
  const int img  = blockIdx.y;
  const int tileBase = (blockIdx.x * 8 + wave) * 16;
  const _Float16* xImg = xLvl + (size_t)img * HW * 64;

  const int m   = lane & 15;           // A-row (pixel) / B-col / D-col
  const int hi  = lane >> 4;           // half-wave selector
  const int kh8 = hi << 3;
  const int p   = tileBase + m;
  const int h   = p >> log2W;
  const int w   = p & (W - 1);

  v8f acc[4] = {};                      // 16 px x 64 out-channels (4 N-tiles)

  for (int tap = 0; tap < 9; ++tap) {
    const int dh = tap / 3 - 1;
    const int dw = tap % 3 - 1;
    const int hs = h + dh, ws2 = w + dw;
    const bool ok = ((unsigned)hs < (unsigned)W) && ((unsigned)ws2 < (unsigned)W);
    const _Float16* arow = xImg + (size_t)(ok ? (hs * W + ws2) : 0) * 64;
    const _Float16* bbase = Bprep + tap * 4096 + m * 64 + (hi << 4);
#pragma unroll
    for (int kc = 0; kc < 2; ++kc) {
      v16h a;
      if (ok) {
        v8h c0 = *(const v8h*)(arow + kc * 32 + (hi << 4));
        v8h c1 = *(const v8h*)(arow + kc * 32 + (hi << 4) + 8);
#pragma unroll
        for (int i = 0; i < 8; ++i) { a[i] = c0[i]; a[i + 8] = c1[i]; }
      } else {
#pragma unroll
        for (int i = 0; i < 16; ++i) a[i] = (_Float16)0.f;
      }
#pragma unroll
      for (int t = 0; t < 4; ++t) {
        v16h b = *(const v16h*)(bbase + t * 1024 + kc * 32);
        acc[t] = __builtin_amdgcn_wmma_f32_16x16x32_f16(false, a, false, b,
                                                        (short)0, acc[t], false, false);
      }
    }
  }

  // epilogue: y = acc + conv_b -> LDS (f16), then head WMMA (cls|bbx)
  _Float16* ytp = yTile[wave];
#pragma unroll
  for (int t = 0; t < 4; ++t) {
    const int n = m + (t << 4);
    const float cb = conv_b[n];
#pragma unroll
    for (int r = 0; r < 8; ++r) {
      const int mm = r + (hi << 3);
      ytp[mm * 64 + n] = (_Float16)(acc[t][r] + cb);
    }
  }
  asm volatile("s_wait_dscnt 0x0" ::: "memory");  // within-wave LDS store->load ordering

  const _Float16* bhb = Bh + m * 64 + (hi << 4);
  v16h hb0 = *(const v16h*)(bhb);
  v16h hb1 = *(const v16h*)(bhb + 32);
  v16h ya0, ya1;
  {
    const _Float16* yr = ytp + m * 64;
    v8h c0 = *(const v8h*)(yr + (hi << 4));
    v8h c1 = *(const v8h*)(yr + (hi << 4) + 8);
    v8h c2 = *(const v8h*)(yr + 32 + (hi << 4));
    v8h c3 = *(const v8h*)(yr + 32 + (hi << 4) + 8);
#pragma unroll
    for (int i = 0; i < 8; ++i) { ya0[i] = c0[i]; ya0[i + 8] = c1[i];
                                  ya1[i] = c2[i]; ya1[i + 8] = c3[i]; }
  }
  v8f d = {};
  d = __builtin_amdgcn_wmma_f32_16x16x32_f16(false, ya0, false, hb0, (short)0, d, false, false);
  d = __builtin_amdgcn_wmma_f32_16x16x32_f16(false, ya1, false, hb1, (short)0, d, false, false);

  const int col = m;
  const bool isCls = col < 3;
  const bool isReg = (col >= 3) && (col < 15);
  float hb = 0.f;
  if (isCls) hb = cls_b[col];
  else if (isReg) hb = bbx_b[col - 3];
  float* scI = scores + (size_t)img * N_ANCH;
  float* deI = deltas + (size_t)img * N_ANCH * 4;
#pragma unroll
  for (int r = 0; r < 8; ++r) {
    const int mm = r + (hi << 3);
    const int ai = lvlOff + (tileBase + mm) * 3;
    const float v = d[r] + hb;
    if (isCls) scI[ai + col] = v;
    else if (isReg) {
      const int c = col - 3;
      deI[(size_t)(ai + (c >> 2)) * 4 + (c & 3)] = v;
    }
  }
}

// ---------------- exact per-level top-1000 (radix select, JAX tie order) ----------------
__global__ __launch_bounds__(1024) void k_select(const float* __restrict__ scores,
                                                 int* __restrict__ selIdx,
                                                 float* __restrict__ selScore) {
  const int lvl = blockIdx.x, img = blockIdx.y;
  const int n   = c_lvlCnt[lvl];
  const int off = c_lvlOff[lvl];
  const float* s = scores + (size_t)img * N_ANCH + off;
  const int tid = threadIdx.x;

  __shared__ unsigned hist[256];
  __shared__ unsigned scan[1024];
  __shared__ unsigned sPrefix, sRemain;

  if (tid == 0) { sPrefix = 0u; sRemain = 1000u; }
  __syncthreads();

  for (int pass = 3; pass >= 0; --pass) {
    if (tid < 256) hist[tid] = 0u;
    __syncthreads();
    const unsigned prefix = sPrefix;
    const unsigned hiMask = (pass == 3) ? 0u : (0xFFFFFFFFu << ((pass + 1) * 8));
    for (int i = tid; i < n; i += 1024) {
      const unsigned k = sortKey(s[i]);
      if ((k & hiMask) == prefix)
        atomicAdd(&hist[(k >> (pass * 8)) & 255u], 1u);
    }
    __syncthreads();
    if (tid == 0) {
      unsigned rem = sRemain, cum = 0u;
      int b = 255;
      for (; b > 0; --b) {
        if (cum + hist[b] >= rem) break;
        cum += hist[b];
      }
      sPrefix = prefix | ((unsigned)b << (pass * 8));
      sRemain = rem - cum;
    }
    __syncthreads();
  }
  const unsigned T = sPrefix;

  // ordered compaction: strictly-greater, then equals by ascending index
  const int chunk = n >> 10;               // all level counts divisible by 1024
  const int beg = tid * chunk, end = beg + chunk;
  unsigned cntG = 0u, cntE = 0u;
  for (int i = beg; i < end; ++i) {
    const unsigned k = sortKey(s[i]);
    cntG += (k > T);
    cntE += (k == T);
  }
  scan[tid] = cntG; __syncthreads();
  for (int o = 1; o < 1024; o <<= 1) {
    unsigned v = (tid >= o) ? scan[tid - o] : 0u;
    __syncthreads(); scan[tid] += v; __syncthreads();
  }
  const unsigned gEx = scan[tid] - cntG;
  const unsigned Gtot = scan[1023];
  __syncthreads();
  scan[tid] = cntE; __syncthreads();
  for (int o = 1; o < 1024; o <<= 1) {
    unsigned v = (tid >= o) ? scan[tid - o] : 0u;
    __syncthreads(); scan[tid] += v; __syncthreads();
  }
  const unsigned eEx = scan[tid] - cntE;
  const unsigned E = 1000u - Gtot;

  int* oIdx = selIdx + img * 3000 + lvl * 1000;
  float* oSc = selScore + img * 3000 + lvl * 1000;
  unsigned pg = gEx, pe = eEx;
  for (int i = beg; i < end; ++i) {
    const float v = s[i];
    const unsigned k = sortKey(v);
    if (k > T) { oIdx[pg] = off + i; oSc[pg] = v; ++pg; }
    else if (k == T && pe < E) { oIdx[Gtot + pe] = off + i; oSc[Gtot + pe] = v; ++pe; }
  }
}

// ---------------- decode + stable sort + NMS + post-NMS top-k ----------------
__global__ __launch_bounds__(1024) void k_final(
    const int* __restrict__ selIdx, const float* __restrict__ selScore,
    const float* __restrict__ deltas,
    float* __restrict__ slotBoff, float* __restrict__ slotProb, int* __restrict__ slotValid,
    float* __restrict__ out) {
  const int img = blockIdx.x;
  const int tid = threadIdx.x;
  __shared__ unsigned long long skey[4096];
  __shared__ unsigned short spay[4096];
  __shared__ unsigned char keepArr[3000];
  __shared__ float bx[4];
  __shared__ int bkeep;

  const int* sIdx = selIdx + img * 3000;
  const float* sSc = selScore + img * 3000;
  float* boffW = slotBoff + img * 3000 * 4;
  float* probW = slotProb + img * 3000;
  int*   valW  = slotValid + img * 3000;

  for (int slot = tid; slot < 4096; slot += 1024) {
    unsigned long long key = 0ull;
    if (slot < 3000) {
      const int lvl = (slot >= 2000) ? 2 : ((slot >= 1000) ? 1 : 0);
      const int ai = sIdx[slot];
      const float sc = sSc[slot];
      const int local = ai - c_lvlOff[lvl];
      const int q = local / 3;
      const int a = local - q * 3;
      const int Wl = c_lvlW[lvl];
      const int hh = q >> c_lvlLog2W[lvl];
      const int ww = q & (Wl - 1);
      const float stride = c_lvlStride[lvl];
      const float cxa = hh * stride, cya = ww * stride;
      const float wa = c_wa[lvl][a], ha = c_ha[lvl][a];
      const float* dp = deltas + ((size_t)img * N_ANCH + ai) * 4;
      float dx = dp[0], dy = dp[1], dwv = dp[2], dhv = dp[3];
      const float CLIP = 4.135166556742356f;   // log(1000/16)
      dwv = fminf(dwv, CLIP);
      dhv = fminf(dhv, CLIP);
      const float cx = dx * wa + cxa;
      const float cy = dy * ha + cya;
      const float bw = __expf(dwv) * wa;
      const float bh = __expf(dhv) * ha;
      const float x0 = fminf(fmaxf(cx - 0.5f * bw, 0.f), 2048.f);
      const float y0 = fminf(fmaxf(cy - 0.5f * bh, 0.f), 2048.f);
      const float x1 = fminf(fmaxf(cx + 0.5f * bw, 0.f), 2048.f);
      const float y1 = fminf(fmaxf(cy + 0.5f * bh, 0.f), 2048.f);
      const float prob = 1.f / (1.f + __expf(-sc));
      const int valid = ((x1 - x0) >= 0.001f) && ((y1 - y0) >= 0.001f) && (prob > 0.f);
      const float lo = lvl * 4096.f;
      boffW[slot * 4 + 0] = x0 + lo;
      boffW[slot * 4 + 1] = y0 + lo;
      boffW[slot * 4 + 2] = x1 + lo;
      boffW[slot * 4 + 3] = y1 + lo;
      probW[slot] = prob;
      valW[slot] = valid;
      const unsigned pk = __float_as_uint(prob) | 0x80000000u;  // prob in (0,1)
      key = valid ? ((((unsigned long long)pk) << 32) | (unsigned long long)(0xFFFFFFFFu - (unsigned)slot))
                  : (unsigned long long)(0xFFFFFFFFu - (unsigned)slot);
    }
    skey[slot] = key;
    spay[slot] = (unsigned short)slot;
  }
  __syncthreads();

  // bitonic sort, descending, 4096 entries
  for (int k = 2; k <= 4096; k <<= 1) {
    for (int j = k >> 1; j > 0; j >>= 1) {
      for (int idx = tid; idx < 4096; idx += 1024) {
        const int ixj = idx ^ j;
        if (ixj > idx) {
          const unsigned long long a = skey[idx], b = skey[ixj];
          const bool desc = ((idx & k) == 0);
          if (desc ? (a < b) : (a > b)) {
            skey[idx] = b; skey[ixj] = a;
            const unsigned short t = spay[idx]; spay[idx] = spay[ixj]; spay[ixj] = t;
          }
        }
      }
      __syncthreads();
    }
  }

  // gather owned candidates (thread tid owns sorted positions tid, tid+1024, tid+2048)
  float obx[3][4];
  int   oval[3];
  bool  sup[3];
#pragma unroll
  for (int mo = 0; mo < 3; ++mo) {
    const int jj = tid + (mo << 10);
    sup[mo] = false;
    if (jj < 3000) {
      const int slot = spay[jj];
      obx[mo][0] = boffW[slot * 4 + 0];
      obx[mo][1] = boffW[slot * 4 + 1];
      obx[mo][2] = boffW[slot * 4 + 2];
      obx[mo][3] = boffW[slot * 4 + 3];
      oval[mo] = valW[slot];
    } else {
      obx[mo][0] = obx[mo][1] = obx[mo][2] = obx[mo][3] = 0.f;
      oval[mo] = 0;
    }
  }
  __syncthreads();

  // greedy NMS (sequential over sorted order, parallel suppression)
  for (int i = 0; i < 3000; ++i) {
    const int owner = i & 1023;
    const int mo = i >> 10;
    if (tid == owner) {
      const int kp = (oval[mo] && !sup[mo]) ? 1 : 0;
      keepArr[i] = (unsigned char)kp;
      bkeep = kp;
      if (kp) { bx[0] = obx[mo][0]; bx[1] = obx[mo][1]; bx[2] = obx[mo][2]; bx[3] = obx[mo][3]; }
    }
    __syncthreads();
    if (bkeep) {
      const float ax0 = bx[0], ay0 = bx[1], ax1 = bx[2], ay1 = bx[3];
      const float areaA = (ax1 - ax0) * (ay1 - ay0);
#pragma unroll
      for (int m2 = 0; m2 < 3; ++m2) {
        const int jj = tid + (m2 << 10);
        if (jj > i && jj < 3000 && !sup[m2]) {
          const float ix0 = fmaxf(ax0, obx[m2][0]);
          const float iy0 = fmaxf(ay0, obx[m2][1]);
          const float ix1 = fminf(ax1, obx[m2][2]);
          const float iy1 = fminf(ay1, obx[m2][3]);
          const float iw = fmaxf(ix1 - ix0, 0.f);
          const float ih = fmaxf(iy1 - iy0, 0.f);
          const float inter = iw * ih;
          const float areaB = (obx[m2][2] - obx[m2][0]) * (obx[m2][3] - obx[m2][1]);
          const float iou = inter / (areaA + areaB - inter + 1e-9f);
          if (iou > 0.7f) sup[m2] = true;
        }
      }
    }
    __syncthreads();
  }

  // post-NMS top-1000: kept in order, then rejected fillers (score -1)
  if (tid == 0) {
    int outPos = 0;
    for (int pass = 0; pass < 2 && outPos < 1000; ++pass) {
      for (int i = 0; i < 3000 && outPos < 1000; ++i) {
        const int kp = keepArr[i];
        if ((pass == 0) != (kp != 0)) continue;
        const int slot = spay[i];
        const int lvl = (slot >= 2000) ? 2 : ((slot >= 1000) ? 1 : 0);
        const float lo = lvl * 4096.f;
        float* ob = out + ((size_t)img * 1000 + outPos) * 4;
        ob[0] = boffW[slot * 4 + 0] - lo;
        ob[1] = boffW[slot * 4 + 1] - lo;
        ob[2] = boffW[slot * 4 + 2] - lo;
        ob[3] = boffW[slot * 4 + 3] - lo;
        out[8000 + img * 1000 + outPos] = (pass == 0) ? probW[slot] : -1.0f;
        ++outPos;
      }
    }
  }
}

extern "C" void kernel_launch(void* const* d_in, const int* in_sizes, int n_in,
                              void* d_out, int out_size, void* d_ws, size_t ws_size,
                              hipStream_t stream) {
  (void)in_sizes; (void)n_in; (void)out_size; (void)ws_size;
  const float* f0     = (const float*)d_in[1];
  const float* f1     = (const float*)d_in[2];
  const float* f2     = (const float*)d_in[3];
  const float* conv_w = (const float*)d_in[4];
  const float* conv_b = (const float*)d_in[5];
  const float* cls_w  = (const float*)d_in[6];
  const float* cls_b  = (const float*)d_in[7];
  const float* bbx_w  = (const float*)d_in[8];
  const float* bbx_b  = (const float*)d_in[9];
  float* out = (float*)d_out;

  char* ws = (char*)d_ws;
  size_t off = 0;
  auto carve = [&](size_t bytes) -> char* {
    off = (off + 255) & ~(size_t)255;
    char* p = ws + off;
    off += bytes;
    return p;
  };

  _Float16* x0 = (_Float16*)carve((size_t)2 * 65536 * 64 * 2);
  _Float16* x1 = (_Float16*)carve((size_t)2 * 16384 * 64 * 2);
  _Float16* x2 = (_Float16*)carve((size_t)2 * 4096 * 64 * 2);
  _Float16* Bprep = (_Float16*)carve((size_t)9 * 4096 * 2);
  _Float16* Bh    = (_Float16*)carve((size_t)1024 * 2);
  float* scores   = (float*)carve((size_t)2 * N_ANCH * 4);
  float* deltas   = (float*)carve((size_t)2 * N_ANCH * 16);
  int*   selIdx   = (int*)carve((size_t)2 * 3000 * 4);
  float* selScore = (float*)carve((size_t)2 * 3000 * 4);
  float* slotBoff = (float*)carve((size_t)2 * 3000 * 16);
  float* slotProb = (float*)carve((size_t)2 * 3000 * 4);
  int*   slotValid= (int*)carve((size_t)2 * 3000 * 4);

  k_prep<<<dim3(40), dim3(256), 0, stream>>>(conv_w, cls_w, bbx_w, Bprep, Bh);
  k_nhwc<<<dim3(2048), dim3(256), 0, stream>>>(f0, x0, 65536, 16, 2 * 64 * 65536);
  k_nhwc<<<dim3(1024), dim3(256), 0, stream>>>(f1, x1, 16384, 14, 2 * 64 * 16384);
  k_nhwc<<<dim3(512),  dim3(256), 0, stream>>>(f2, x2, 4096, 12, 2 * 64 * 4096);

  k_conv_head<<<dim3(512, 2), dim3(256), 0, stream>>>(x0, Bprep, Bh, conv_b, cls_b, bbx_b,
                                                      scores, deltas, 256, 8, 65536, 0);
  k_conv_head<<<dim3(128, 2), dim3(256), 0, stream>>>(x1, Bprep, Bh, conv_b, cls_b, bbx_b,
                                                      scores, deltas, 128, 7, 16384, 196608);
  k_conv_head<<<dim3(32, 2),  dim3(256), 0, stream>>>(x2, Bprep, Bh, conv_b, cls_b, bbx_b,
                                                      scores, deltas, 64, 6, 4096, 245760);

  k_select<<<dim3(3, 2), dim3(1024), 0, stream>>>(scores, selIdx, selScore);
  k_final<<<dim3(2), dim3(1024), 0, stream>>>(selIdx, selScore, deltas,
                                              slotBoff, slotProb, slotValid, out);
}
